// MultiHeadAttention_82051055223285
// MI455X (gfx1250) — matrix operand order repaired and checked by reading, not compiled, not run on hardware
//
#include <hip/hip_runtime.h>

// MI455X (gfx1250) fused multi-head attention with learned KV/Q projections.
// bf16 WMMA (v_wmma_f32_16x16x32_bf16) for all GEMMs; KV projection fused with
// attention so K/V never touch HBM; TDM (tensor_load_to_lds) prefetches the
// next input tile into LDS while WMMAs run on the current one. The projection
// GEMM holds all 16 Wkv B-fragments in VGPRs per head and double-buffers the
// A-fragment DS loads so LDS latency overlaps WMMA execution.

#define S_ 4096
#define Q_ 32
#define B_ 8
#define IDIM 512
#define NHEAD 8
#define HDIM 64
#define S_TILE 64
#define TILES_PER_BLK 2

typedef __attribute__((ext_vector_type(16))) __bf16 v16bf;
typedef __attribute__((ext_vector_type(8)))  float  v8f;
typedef unsigned int u32x4 __attribute__((ext_vector_type(4)));
typedef int          i32x4 __attribute__((ext_vector_type(4)));
typedef int          i32x8 __attribute__((ext_vector_type(8)));

union Frag {
  v16bf bf;
  uint4 u[2];
};

__device__ __forceinline__ unsigned short f32_bf16(float f) {
  unsigned int u = __float_as_uint(f);
  unsigned int r = (u + 0x7FFFu + ((u >> 16) & 1u)) >> 16;
  return (unsigned short)r;
}

// ---- Tensor Data Mover: 2-D f32 tile (rows x cols, row-major, stride=cols)
// from global into LDS at byte offset lds_off. D# per CDNA5 ISA ch.8.
__device__ __forceinline__ void tdm_load_tile_f32(const float* gptr, unsigned lds_off,
                                                  int rows, int cols) {
  unsigned long long ga = (unsigned long long)(uintptr_t)gptr;
  u32x4 g0;
  g0[0] = 1u;                                   // count=1, user descriptor
  g0[1] = lds_off;                              // lds_addr [63:32]
  g0[2] = (unsigned)ga;                         // global_addr low  [95:64]
  g0[3] = (unsigned)(ga >> 32) | 0x80000000u;   // addr hi [120:96], type=2 [127:126]
  i32x8 g1;
  g1[0] = 0x20000;                              // data_size=2 (4 bytes)
  g1[1] = (cols & 0xFFFF) << 16;                // tensor_dim0 [79:48] low half
  g1[2] = ((unsigned)cols >> 16) | ((rows & 0xFFFF) << 16);  // dim0 hi | dim1 lo
  g1[3] = (cols & 0xFFFF) << 16;                // dim1 hi (=0) | tile_dim0 [127:112]
  g1[4] = rows & 0xFFFF;                        // tile_dim1 | tile_dim2=0
  g1[5] = cols;                                 // tensor_dim0_stride low32
  g1[6] = 0;
  g1[7] = 0;
  i32x4 z4 = {0, 0, 0, 0};
#if __clang_major__ >= 23
  i32x8 z8 = {0, 0, 0, 0, 0, 0, 0, 0};
  __builtin_amdgcn_tensor_load_to_lds(g0, g1, z4, z4, z8, 0);
#else
  __builtin_amdgcn_tensor_load_to_lds(g0, g1, z4, z4, 0);
#endif
}

__global__ void k_zero(float* __restrict__ p, int n) {
  int i = blockIdx.x * blockDim.x + threadIdx.x;
  if (i < n) p[i] = 0.f;
}

__global__ void k_cvt_bf16(const float* __restrict__ src, unsigned short* __restrict__ dst, int n) {
  int i = blockIdx.x * blockDim.x + threadIdx.x;
  if (i < n) dst[i] = f32_bf16(src[i]);
}

// q_proj[b*32+q][o] = bq[o] + sum_c queries[b,q,c] * Wq[o,c]   (bf16 out)
__global__ void k_qproj(const float* __restrict__ queries, const float* __restrict__ Wq,
                        const float* __restrict__ bq, unsigned short* __restrict__ qp) {
  int o = blockIdx.x * blockDim.x + threadIdx.x;
  if (o >= B_ * Q_ * IDIM) return;
  int col = o & (IDIM - 1);
  int row = o >> 9;
  const float4* qr = (const float4*)(queries + (size_t)row * IDIM);
  const float4* wr = (const float4*)(Wq + (size_t)col * IDIM);
  float acc = bq[col];
  for (int i = 0; i < IDIM / 4; ++i) {
    float4 a = qr[i], w = wr[i];
    acc += a.x * w.x + a.y * w.y + a.z * w.z + a.w * w.w;
  }
  qp[o] = f32_bf16(acc);
}

// Main fused kernel: one block per (batch, 2 consecutive 64-row S tiles).
__launch_bounds__(256)
__global__ void k_attn(const float* __restrict__ inputs, const float* __restrict__ masks,
                       const float* __restrict__ bkv,
                       const unsigned short* __restrict__ wkv16,
                       const unsigned short* __restrict__ qp16,
                       float* __restrict__ acc) {
  // Single shared block -> Xf sits at LDS byte offset 0 (TDM target).
  __shared__ struct {
    float          Xf[S_TILE * IDIM];  // 128 KB  f32 staging (TDM destination)
    unsigned short Xs[S_TILE * 520];   //  65 KB  inputs tile [s][c] bf16
    unsigned short Ks[S_TILE * 72];    //   9 KB  k tile [s][d] bf16
    unsigned short Vt[80 * 72];        //  11 KB  v tile^T [d][s]; rows 64..79 ones/zeros
    unsigned short Wb[Q_ * 72];        //   4 KB  exp-scores [q][s] bf16
  } L;

  const int b    = blockIdx.x >> 4;              // 16 blocks per batch
  const int t0   = (blockIdx.x & 15) * TILES_PER_BLK;
  const int tid  = threadIdx.x;
  const int wave = tid >> 5;
  const int lane = tid & 31;
  const int lhi  = lane >> 4;
  const int llo  = lane & 15;

  // init Vt extension rows (d=64 -> ones column = softmax denominator)
  for (int idx = tid; idx < 16 * S_TILE; idx += 256) {
    int d = 64 + (idx >> 6);
    int s = idx & 63;
    L.Vt[d * 72 + s] = (d == 64) ? (unsigned short)0x3F80 : (unsigned short)0;
  }

  // TDM prefetch of first tile (wave 0; EXEC-independent, TENSORcnt-tracked)
  if (wave == 0) {
    tdm_load_tile_f32(inputs + ((size_t)b * S_ + t0 * S_TILE) * IDIM, 0, S_TILE, IDIM);
  }

  const float inv_sqrt = 0.044194173824159216f; // 1/sqrt(512)

  for (int tt = 0; tt < TILES_PER_BLK; ++tt) {
    const int s0 = (t0 + tt) * S_TILE;

    if (wave == 0) __builtin_amdgcn_s_wait_tensorcnt(0);
    __syncthreads();

    // ---- convert staged f32 tile -> bf16 compute tile ----
    for (int idx = tid; idx < (S_TILE * IDIM) / 4; idx += 256) {
      float4 v = *(const float4*)&L.Xf[idx * 4];
      int f = idx * 4;
      int row = f >> 9, col = f & (IDIM - 1);
      unsigned short* d = &L.Xs[row * 520 + col];
      d[0] = f32_bf16(v.x); d[1] = f32_bf16(v.y); d[2] = f32_bf16(v.z); d[3] = f32_bf16(v.w);
    }
    __syncthreads();

    // ---- kick TDM prefetch of the next tile; overlaps with head loop ----
    if (wave == 0 && tt + 1 < TILES_PER_BLK) {
      tdm_load_tile_f32(inputs + ((size_t)b * S_ + (t0 + tt + 1) * S_TILE) * IDIM,
                        0, S_TILE, IDIM);
    }

    for (int h = 0; h < NHEAD; ++h) {
      // ---- (a) KV projection: [64 s] x [128 out (k|v)], K = 512 ----
      // tn is wave-constant -> keep all 16 Wkv B-fragments live in VGPRs and
      // reuse them across the 4 M-tiles; double-buffer the A fragments.
      {
        const int tn   = wave & 7;
        const int nb   = tn * 16;
        const int ocol = nb + llo;                    // 0..127 (k cols | v cols)
        const int wr   = (ocol < 64) ? (h * HDIM + ocol)
                                     : (IDIM + h * HDIM + (ocol - 64));
        const float bias = bkv[wr];

        Frag Bf[16];
        #pragma unroll
        for (int kk = 0; kk < 16; ++kk) {
          const uint4* p = (const uint4*)&wkv16[(size_t)wr * IDIM + kk * 32 + (lhi << 3)];
          Bf[kk].u[0] = p[0]; Bf[kk].u[1] = p[2];
        }

        #pragma unroll
        for (int tm = 0; tm < 4; ++tm) {
          const int mb = tm * 16;
          const unsigned short* xrow = &L.Xs[(mb + llo) * 520];
          v8f c = {};
          Frag a0;
          { const uint4* p = (const uint4*)&xrow[lhi << 3];
            a0.u[0] = p[0]; a0.u[1] = p[2]; }
          #pragma unroll
          for (int kk = 0; kk < 16; ++kk) {
            Frag an = a0;
            if (kk + 1 < 16) {              // prefetch next A before this WMMA
              const uint4* p = (const uint4*)&xrow[(kk + 1) * 32 + (lhi << 3)];
              an.u[0] = p[0]; an.u[1] = p[2];
            }
            c = __builtin_amdgcn_wmma_f32_16x16x32_bf16(false, a0.bf, false, Bf[kk].bf,
                                                        (short)0, c, false, false);
            a0 = an;
          }
          if (ocol < 64) {                  // k -> natural [s][d]
            const int d = ocol;
            #pragma unroll
            for (int i = 0; i < 8; ++i) {
              int s = mb + i + (lhi << 3);
              L.Ks[s * 72 + d] = f32_bf16(c[i] + bias);
            }
          } else {                          // v -> transposed [d][s]
            const int d = ocol - 64;
            union { unsigned short h[8]; uint4 u; } pk;
            #pragma unroll
            for (int i = 0; i < 8; ++i) pk.h[i] = f32_bf16(c[i] + bias);
            int sb = mb + (lhi << 3);
            *(uint4*)&L.Vt[d * 72 + sb] = pk.u;
          }
        }
      }
      __syncthreads();

      // ---- (b) scores U = q @ k^T (32x64, K=64), then elu-clip-exp-mask ----
      {
        int qb = (wave >> 2) * 16, sb = (wave & 3) * 16;
        Frag a0, a1, b0, b1;
        { const int row = b * Q_ + qb + llo;
          const uint4* p = (const uint4*)&qp16[(size_t)row * IDIM + h * HDIM + (lhi << 4)];
          a0.u[0] = p[0]; a0.u[1] = p[1];
          const uint4* q2 = (const uint4*)&qp16[(size_t)row * IDIM + h * HDIM + 32 + (lhi << 4)];
          a1.u[0] = q2[0]; a1.u[1] = q2[1]; }
        { const int s = sb + llo;
          const uint4* p = (const uint4*)&L.Ks[s * 72 + (lhi << 4)];
          b0.u[0] = p[0]; b0.u[1] = p[1];
          const uint4* q2 = (const uint4*)&L.Ks[s * 72 + 32 + (lhi << 4)];
          b1.u[0] = q2[0]; b1.u[1] = q2[1]; }
        v8f c = {};
        c = __builtin_amdgcn_wmma_f32_16x16x32_bf16(false, a0.bf, false, b0.bf,
                                                    (short)0, c, false, false);
        c = __builtin_amdgcn_wmma_f32_16x16x32_bf16(false, a1.bf, false, b1.bf,
                                                    (short)0, c, false, false);
        int s_l = sb + llo;
        float m = masks[(size_t)b * S_ + s0 + s_l];
        #pragma unroll
        for (int i = 0; i < 8; ++i) {
          float u = c[i] * inv_sqrt;
          float e = (u > 0.f) ? u : (__expf(u) - 1.f);
          e = fminf(fmaxf(e, -15.f), 15.f);
          float w = __expf(e) * m;
          int qrow = qb + i + (lhi << 3);
          L.Wb[qrow * 72 + s_l] = f32_bf16(w);
        }
      }
      __syncthreads();

      // ---- (c) acc += W @ [V | ones] (32x80, K=64); col 64 = denominator ----
      for (int t = wave; t < 10; t += 8) {
        int qb = (t / 5) * 16, db = (t % 5) * 16;
        Frag a0, a1, b0, b1;
        { const int row = qb + llo;
          const uint4* p = (const uint4*)&L.Wb[row * 72 + (lhi << 4)];
          a0.u[0] = p[0]; a0.u[1] = p[1];
          const uint4* q2 = (const uint4*)&L.Wb[row * 72 + 32 + (lhi << 4)];
          a1.u[0] = q2[0]; a1.u[1] = q2[1]; }
        { const int d = db + llo;
          const uint4* p = (const uint4*)&L.Vt[d * 72 + (lhi << 4)];
          b0.u[0] = p[0]; b0.u[1] = p[1];
          const uint4* q2 = (const uint4*)&L.Vt[d * 72 + 32 + (lhi << 4)];
          b1.u[0] = q2[0]; b1.u[1] = q2[1]; }
        v8f c = {};
        c = __builtin_amdgcn_wmma_f32_16x16x32_bf16(false, a0.bf, false, b0.bf,
                                                    (short)0, c, false, false);
        c = __builtin_amdgcn_wmma_f32_16x16x32_bf16(false, a1.bf, false, b1.bf,
                                                    (short)0, c, false, false);
        int col = db + llo;
        #pragma unroll
        for (int i = 0; i < 8; ++i) {
          int q = qb + i + (lhi << 3);
          atomicAdd(&acc[(((size_t)b * NHEAD + h) * Q_ + q) * 80 + col], c[i]);
        }
      }
      __syncthreads();
    }
  }
}

__global__ void k_final(const float* __restrict__ acc, float* __restrict__ out) {
  int i = blockIdx.x * blockDim.x + threadIdx.x;   // b*32*512 + q*512 + h*64 + d
  if (i >= B_ * Q_ * IDIM) return;
  int d = i & 63;
  int h = (i >> 6) & 7;
  int q = (i >> 9) & 31;
  int b = i >> 14;
  const float* base = &acc[(((size_t)b * NHEAD + h) * Q_ + q) * 80];
  out[i] = base[d] / base[64];
}

extern "C" void kernel_launch(void* const* d_in, const int* in_sizes, int n_in,
                              void* d_out, int out_size, void* d_ws, size_t ws_size,
                              hipStream_t stream) {
  const float* inputs  = (const float*)d_in[0];
  const float* queries = (const float*)d_in[1];
  const float* masks   = (const float*)d_in[2];
  const float* Wkv     = (const float*)d_in[3];
  const float* bkv     = (const float*)d_in[4];
  const float* Wq      = (const float*)d_in[5];
  const float* bq      = (const float*)d_in[6];
  float* out = (float*)d_out;

  // workspace: [Wkv bf16 : 1MB][q_proj bf16 : 256KB][acc f32 : 640KB]
  char* ws = (char*)d_ws;
  unsigned short* wkv16 = (unsigned short*)ws;
  unsigned short* qp16  = (unsigned short*)(ws + (1u << 20));
  float*          acc   = (float*)(ws + (1u << 20) + (256u << 10));

  const int n_acc = B_ * NHEAD * Q_ * 80;
  const int n_wkv = 2 * IDIM * IDIM;
  const int n_qp  = B_ * Q_ * IDIM;

  k_zero    <<<(n_acc + 255) / 256, 256, 0, stream>>>(acc, n_acc);
  k_cvt_bf16<<<(n_wkv + 255) / 256, 256, 0, stream>>>(Wkv, wkv16, n_wkv);
  k_qproj   <<<(n_qp  + 255) / 256, 256, 0, stream>>>(queries, Wq, bq, qp16);
  k_attn    <<<B_ * (S_ / S_TILE) / TILES_PER_BLK, 256, 0, stream>>>
            (inputs, masks, bkv, wkv16, qp16, acc);
  k_final   <<<(n_qp  + 255) / 256, 256, 0, stream>>>(acc, out);
}
